// EncoderGAT_25185688224513
// MI455X (gfx1250) — matrix operand order repaired and checked by reading, not compiled, not run on hardware
//
#include <hip/hip_runtime.h>
#include <hip/hip_bf16.h>
#include <cstdint>
#include <cstddef>

typedef __attribute__((ext_vector_type(16))) _Float16 v16h;
typedef __attribute__((ext_vector_type(8)))  _Float16 v8h;
typedef __attribute__((ext_vector_type(8)))  float    v8f;

#define IN_CH   512
#define H1DIM   64      // HEADS * C1
#define NHEADS  8
#define C1DIM   8
#define OUTC    128
#define NEG_SLOPE 0.2f

// ---------- device helpers ----------
static __device__ __forceinline__ int fenc(float f) {
  int i = __float_as_int(f);
  return i >= 0 ? i : (i ^ 0x7fffffff);   // order-preserving int encoding
}
static __device__ __forceinline__ float fdec(int i) {
  return __int_as_float(i >= 0 ? i : (i ^ 0x7fffffff));
}
static __device__ __forceinline__ float lrelu(float a) {
  return a > 0.f ? a : NEG_SLOPE * a;
}
static __device__ __forceinline__ float eluf(float a) {
  return a > 0.f ? a : (__expf(a) - 1.f);
}
static __device__ __forceinline__ void edge_sd(const int* __restrict__ ei, int E,
                                               int e, int& src, int& dst) {
  if (e < E) { src = ei[e]; dst = ei[E + e]; }
  else       { src = e - E; dst = e - E; }   // self-loops appended
}

// ---------- weight swizzle: f32 row-major [K,Ncols] -> WMMA B-fragment layout f16 ----------
// Bs[((kt*NT + nt)*32 + lane)*16 + v] = B[kt*32 + ((lane>>4)<<3) + (v&7) + ((v>>3)<<4), nt*16 + (lane&15)]
__global__ void k_swz_b(const float* __restrict__ B, _Float16* __restrict__ Bs,
                        int K, int Ncols) {
  int i = blockIdx.x * blockDim.x + threadIdx.x;
  if (i >= K * Ncols) return;
  int v    = i & 15;
  int lane = (i >> 4) & 31;
  int t    = i >> 9;
  int NT   = Ncols >> 4;
  int nt   = t % NT;
  int kt   = t / NT;
  int k = kt * 32 + ((lane >> 4) << 3) + (v & 7) + ((v >> 3) << 4);
  int n = nt * 16 + (lane & 15);
  Bs[i] = (_Float16)B[(size_t)k * Ncols + n];
}

__global__ void k_zero(float* __restrict__ p, int n) {
  int i = blockIdx.x * blockDim.x + threadIdx.x;
  if (i < n) p[i] = 0.f;
}

// ---------- GEMM1: h1[N,64] = x[N,512](f32->f16) @ W1s(f16, pre-swizzled) ----------
__global__ void k_gemm1(const float* __restrict__ X, const _Float16* __restrict__ Bs,
                        float* __restrict__ C, int n) {
  const int lane = threadIdx.x & 31;
  const int wid  = blockIdx.x * (blockDim.x >> 5) + (threadIdx.x >> 5);
  const int Mtiles = (n + 15) >> 4;
  if (wid >= Mtiles) return;                 // wave-uniform: EXEC stays all-1s
  const int m0 = wid << 4;
  int m = m0 + (lane & 15);
  if (m >= n) m = n - 1;                     // clamp duplicate-loads for ragged tail
  const int lo = lane >> 4;                  // 0: lanes 0-15, 1: lanes 16-31
  const float* xr = X + (size_t)m * IN_CH;
  const int nn = lane & 15;

  v8f acc[4] = {};
  for (int kt = 0; kt < IN_CH / 32; ++kt) {
    const int ka = kt * 32 + (lo << 3);
    __builtin_prefetch(xr + ka + 128, 0, 1);          // global_prefetch_b8
    // A fragment 16x32 f16: per-lane halves = A[m, ka..ka+7] and A[m, ka+16..ka+23]
    float4 p0 = *(const float4*)(xr + ka);
    float4 p1 = *(const float4*)(xr + ka + 4);
    float4 q0 = *(const float4*)(xr + ka + 16);
    float4 q1 = *(const float4*)(xr + ka + 20);
    v16h a;
    a[0]  = (_Float16)p0.x; a[1]  = (_Float16)p0.y;
    a[2]  = (_Float16)p0.z; a[3]  = (_Float16)p0.w;
    a[4]  = (_Float16)p1.x; a[5]  = (_Float16)p1.y;
    a[6]  = (_Float16)p1.z; a[7]  = (_Float16)p1.w;
    a[8]  = (_Float16)q0.x; a[9]  = (_Float16)q0.y;
    a[10] = (_Float16)q0.z; a[11] = (_Float16)q0.w;
    a[12] = (_Float16)q1.x; a[13] = (_Float16)q1.y;
    a[14] = (_Float16)q1.z; a[15] = (_Float16)q1.w;

    const _Float16* bb = Bs + ((size_t)(kt * 4) * 32 + lane) * 16;
#pragma unroll
    for (int nt = 0; nt < 4; ++nt) {
      v16h b = *(const v16h*)(bb + (size_t)nt * 32 * 16);   // 32B contiguous per lane
      acc[nt] = __builtin_amdgcn_wmma_f32_16x16x32_f16(
          false, a, false, b, (short)0, acc[nt], false, false);
    }
  }
  // C layout: lanes 0-15 -> rows m0+r, lanes 16-31 -> rows m0+8+r, col = lane&15
  const int rowb = m0 + (lo << 3);
  float* cp = C + (size_t)rowb * H1DIM + nn;
  if (m0 + 16 <= n) {
#pragma unroll
    for (int r = 0; r < 8; ++r) {
      float* cr = cp + (size_t)r * H1DIM;
      cr[0]  = acc[0][r];
      cr[16] = acc[1][r];
      cr[32] = acc[2][r];
      cr[48] = acc[3][r];
    }
  } else {
#pragma unroll
    for (int r = 0; r < 8; ++r) {
      if (rowb + r < n) {
        float* cr = cp + (size_t)r * H1DIM;
        cr[0]  = acc[0][r];
        cr[16] = acc[1][r];
        cr[32] = acc[2][r];
        cr[48] = acc[3][r];
      }
    }
  }
}

// ---------- GEMM2: h2[N,128] = h1p[N,64](f16) @ W2s(f16, pre-swizzled) ----------
__global__ void k_gemm2(const _Float16* __restrict__ A, const _Float16* __restrict__ Bs,
                        float* __restrict__ C, int n) {
  const int lane = threadIdx.x & 31;
  const int wid  = blockIdx.x * (blockDim.x >> 5) + (threadIdx.x >> 5);
  const int Mtiles = (n + 15) >> 4;
  if (wid >= Mtiles) return;
  const int m0 = wid << 4;
  int m = m0 + (lane & 15);
  if (m >= n) m = n - 1;
  const int lo = lane >> 4;
  const _Float16* ar = A + (size_t)m * H1DIM;
  const int nn = lane & 15;

  v8f acc[8] = {};
#pragma unroll
  for (int kt = 0; kt < H1DIM / 32; ++kt) {
    const int ka = kt * 32 + (lo << 3);
    v8h alo = *(const v8h*)(ar + ka);          // 16B vector load
    v8h ahi = *(const v8h*)(ar + ka + 16);     // 16B vector load
    v16h a = __builtin_shufflevector(alo, ahi, 0, 1, 2, 3, 4, 5, 6, 7,
                                     8, 9, 10, 11, 12, 13, 14, 15);
    const _Float16* bb = Bs + ((size_t)(kt * 8) * 32 + lane) * 16;
#pragma unroll
    for (int nt = 0; nt < 8; ++nt) {
      v16h b = *(const v16h*)(bb + (size_t)nt * 32 * 16);
      acc[nt] = __builtin_amdgcn_wmma_f32_16x16x32_f16(
          false, a, false, b, (short)0, acc[nt], false, false);
    }
  }
  const int rowb = m0 + (lo << 3);
  float* cp = C + (size_t)rowb * OUTC + nn;
  if (m0 + 16 <= n) {
#pragma unroll
    for (int r = 0; r < 8; ++r) {
      float* cr = cp + (size_t)r * OUTC;
#pragma unroll
      for (int nt = 0; nt < 8; ++nt) cr[nt * 16] = acc[nt][r];
    }
  } else {
#pragma unroll
    for (int r = 0; r < 8; ++r) {
      if (rowb + r < n) {
        float* cr = cp + (size_t)r * OUTC;
#pragma unroll
        for (int nt = 0; nt < 8; ++nt) cr[nt * 16] = acc[nt][r];
      }
    }
  }
}

// ---------- conv1 attention ----------
__global__ void k_attn1(const float* __restrict__ h1, const float* __restrict__ asrc,
                        const float* __restrict__ adst, float* __restrict__ s1,
                        float* __restrict__ d1, int* __restrict__ amax,
                        float* __restrict__ denom, int n) {
  int t = blockIdx.x * blockDim.x + threadIdx.x;
  if (t >= n * NHEADS) return;
  int node = t >> 3, h = t & 7;
  const float4* hp = (const float4*)(h1 + (size_t)node * H1DIM + h * C1DIM);
  const float4* ap = (const float4*)(asrc + h * C1DIM);
  const float4* dp = (const float4*)(adst + h * C1DIM);
  float4 v0 = hp[0], v1 = hp[1];
  float4 a0 = ap[0], a1 = ap[1];
  float4 b0 = dp[0], b1 = dp[1];
  float s = v0.x * a0.x + v0.y * a0.y + v0.z * a0.z + v0.w * a0.w +
            v1.x * a1.x + v1.y * a1.y + v1.z * a1.z + v1.w * a1.w;
  float d = v0.x * b0.x + v0.y * b0.y + v0.z * b0.z + v0.w * b0.w +
            v1.x * b1.x + v1.y * b1.y + v1.z * b1.z + v1.w * b1.w;
  s1[t] = s; d1[t] = d;
  amax[t] = (int)0x80000000;
  denom[t] = 0.f;
}

__global__ void k_edge1_max(const int* __restrict__ ei, int E, int N,
                            const float* __restrict__ s1, const float* __restrict__ d1,
                            int* __restrict__ amax) {
  int t = blockIdx.x * blockDim.x + threadIdx.x;
  int ET = E + N;
  if (t >= ET * NHEADS) return;
  int e = t >> 3, h = t & 7, src, dst;
  edge_sd(ei, E, e, src, dst);
  float a = lrelu(s1[src * NHEADS + h] + d1[dst * NHEADS + h]);
  atomicMax(&amax[dst * NHEADS + h], fenc(a));
}

__global__ void k_edge1_sum(const int* __restrict__ ei, int E, int N,
                            const float* __restrict__ s1, const float* __restrict__ d1,
                            const int* __restrict__ amax, float* __restrict__ denom) {
  int t = blockIdx.x * blockDim.x + threadIdx.x;
  int ET = E + N;
  if (t >= ET * NHEADS) return;
  int e = t >> 3, h = t & 7, src, dst;
  edge_sd(ei, E, e, src, dst);
  float a  = lrelu(s1[src * NHEADS + h] + d1[dst * NHEADS + h]);
  float ea = __expf(a - fdec(amax[dst * NHEADS + h]));
  atomicAdd(&denom[dst * NHEADS + h], ea);
}

__global__ void k_edge1_agg(const int* __restrict__ ei, int E, int N,
                            const float* __restrict__ s1, const float* __restrict__ d1,
                            const int* __restrict__ amax, const float* __restrict__ denom,
                            const float* __restrict__ h1, float* __restrict__ out1) {
  int t = blockIdx.x * blockDim.x + threadIdx.x;
  int ET = E + N;
  if (t >= ET * NHEADS) return;
  int e = t >> 3, h = t & 7, src, dst;
  edge_sd(ei, E, e, src, dst);
  float a  = lrelu(s1[src * NHEADS + h] + d1[dst * NHEADS + h]);
  float ea = __expf(a - fdec(amax[dst * NHEADS + h]));
  float w  = ea / (denom[dst * NHEADS + h] + 1e-16f);
  const float4* hs = (const float4*)(h1 + (size_t)src * H1DIM + h * C1DIM);
  float4 u0 = hs[0], u1 = hs[1];
  float* od = out1 + (size_t)dst * H1DIM + h * C1DIM;
  atomicAdd(&od[0], u0.x * w); atomicAdd(&od[1], u0.y * w);
  atomicAdd(&od[2], u0.z * w); atomicAdd(&od[3], u0.w * w);
  atomicAdd(&od[4], u1.x * w); atomicAdd(&od[5], u1.y * w);
  atomicAdd(&od[6], u1.z * w); atomicAdd(&od[7], u1.w * w);
}

__global__ void k_elu_bias1(const float* __restrict__ out1, const float* __restrict__ bias,
                            _Float16* __restrict__ h1p, int n) {
  int i = blockIdx.x * blockDim.x + threadIdx.x;
  if (i >= n * H1DIM) return;
  float v = out1[i] + bias[i & (H1DIM - 1)];
  h1p[i] = (_Float16)eluf(v);
}

// ---------- conv2 attention (heads=1, C=128) ----------
__global__ void k_attn2(const float* __restrict__ h2, const float* __restrict__ asrc,
                        const float* __restrict__ adst, float* __restrict__ s2,
                        float* __restrict__ d2, int* __restrict__ amax,
                        float* __restrict__ denom, float* __restrict__ out, int n) {
  int node = blockIdx.x * blockDim.x + threadIdx.x;
  if (node >= n) return;
  const float4* hp = (const float4*)(h2 + (size_t)node * OUTC);
  const float4* ap = (const float4*)asrc;
  const float4* dp = (const float4*)adst;
  float4* op = (float4*)(out + (size_t)node * OUTC);
  float s = 0.f, d = 0.f;
  const float4 z = make_float4(0.f, 0.f, 0.f, 0.f);
#pragma unroll 4
  for (int c = 0; c < OUTC / 4; ++c) {
    float4 v = hp[c], a = ap[c], b = dp[c];
    s += v.x * a.x + v.y * a.y + v.z * a.z + v.w * a.w;
    d += v.x * b.x + v.y * b.y + v.z * b.z + v.w * b.w;
    op[c] = z;                             // zero output accumulator
  }
  s2[node] = s; d2[node] = d;
  amax[node] = (int)0x80000000;
  denom[node] = 0.f;
}

__global__ void k_edge2_max(const int* __restrict__ ei, int E, int N,
                            const float* __restrict__ s2, const float* __restrict__ d2,
                            int* __restrict__ amax) {
  int e = blockIdx.x * blockDim.x + threadIdx.x;
  if (e >= E + N) return;
  int src, dst; edge_sd(ei, E, e, src, dst);
  atomicMax(&amax[dst], fenc(lrelu(s2[src] + d2[dst])));
}

__global__ void k_edge2_sum(const int* __restrict__ ei, int E, int N,
                            const float* __restrict__ s2, const float* __restrict__ d2,
                            const int* __restrict__ amax, float* __restrict__ denom) {
  int e = blockIdx.x * blockDim.x + threadIdx.x;
  if (e >= E + N) return;
  int src, dst; edge_sd(ei, E, e, src, dst);
  float a = lrelu(s2[src] + d2[dst]);
  atomicAdd(&denom[dst], __expf(a - fdec(amax[dst])));
}

__global__ void k_edge2_w(const int* __restrict__ ei, int E, int N,
                          const float* __restrict__ s2, const float* __restrict__ d2,
                          const int* __restrict__ amax, const float* __restrict__ denom,
                          float* __restrict__ wbuf) {
  int e = blockIdx.x * blockDim.x + threadIdx.x;
  if (e >= E + N) return;
  int src, dst; edge_sd(ei, E, e, src, dst);
  float a  = lrelu(s2[src] + d2[dst]);
  float ea = __expf(a - fdec(amax[dst]));
  wbuf[e]  = ea / (denom[dst] + 1e-16f);
}

__global__ void k_edge2_agg(const int* __restrict__ ei, int E, int N,
                            const float* __restrict__ wbuf, const float* __restrict__ h2,
                            float* __restrict__ out) {
  int t = blockIdx.x * blockDim.x + threadIdx.x;
  int ET = E + N;
  if (t >= ET * OUTC) return;
  int e = t >> 7, c = t & (OUTC - 1);
  int src, dst; edge_sd(ei, E, e, src, dst);
  atomicAdd(&out[(size_t)dst * OUTC + c], h2[(size_t)src * OUTC + c] * wbuf[e]);
}

__global__ void k_elu_bias2(float* __restrict__ out, const float* __restrict__ bias, int n) {
  int i = blockIdx.x * blockDim.x + threadIdx.x;
  if (i >= n * OUTC) return;
  out[i] = eluf(out[i] + bias[i & (OUTC - 1)]);
}

// ---------- host ----------
extern "C" void kernel_launch(void* const* d_in, const int* in_sizes, int n_in,
                              void* d_out, int out_size, void* d_ws, size_t ws_size,
                              hipStream_t stream) {
  const float* x   = (const float*)d_in[0];
  const int*   ei  = (const int*)  d_in[1];
  const float* W1  = (const float*)d_in[2];
  const float* as1 = (const float*)d_in[3];
  const float* ad1 = (const float*)d_in[4];
  const float* b1  = (const float*)d_in[5];
  const float* W2  = (const float*)d_in[6];
  const float* as2 = (const float*)d_in[7];
  const float* ad2 = (const float*)d_in[8];
  const float* b2  = (const float*)d_in[9];
  float* out = (float*)d_out;

  const int N  = in_sizes[0] / IN_CH;
  const int E  = in_sizes[1] / 2;
  const int ET = E + N;

  // carve workspace
  char* p = (char*)d_ws;
  auto carve = [&](size_t bytes) -> void* {
    void* r = (void*)p;
    p += (bytes + 255) & ~(size_t)255;
    return r;
  };
  float*    h1     = (float*)   carve((size_t)N * H1DIM * 4);
  float*    out1   = (float*)   carve((size_t)N * H1DIM * 4);
  float*    s1     = (float*)   carve((size_t)N * NHEADS * 4);
  float*    d1     = (float*)   carve((size_t)N * NHEADS * 4);
  int*      amax1  = (int*)     carve((size_t)N * NHEADS * 4);
  float*    den1   = (float*)   carve((size_t)N * NHEADS * 4);
  _Float16* h1p    = (_Float16*)carve((size_t)N * H1DIM * 2);
  float*    h2     = (float*)   carve((size_t)N * OUTC * 4);
  float*    s2     = (float*)   carve((size_t)N * 4);
  float*    d2     = (float*)   carve((size_t)N * 4);
  int*      amax2  = (int*)     carve((size_t)N * 4);
  float*    den2   = (float*)   carve((size_t)N * 4);
  float*    w2buf  = (float*)   carve((size_t)ET * 4);
  _Float16* W1s    = (_Float16*)carve((size_t)IN_CH * H1DIM * 2);
  _Float16* W2s    = (_Float16*)carve((size_t)H1DIM * OUTC * 2);

  const int BS = 256;
  auto nb = [](long long t) { return (int)((t + 255) / 256); };
  const int Mtiles = (N + 15) / 16;
  const int gemmBlocks = (Mtiles + 7) / 8;   // 8 waves per 256-thread block

  // weight f32 -> f16, pre-swizzled to WMMA B-fragment layout
  k_swz_b<<<nb(IN_CH * H1DIM), BS, 0, stream>>>(W1, W1s, IN_CH, H1DIM);
  k_swz_b<<<nb(H1DIM * OUTC), BS, 0, stream>>>(W2, W2s, H1DIM, OUTC);

  // ----- layer 1 -----
  k_gemm1<<<gemmBlocks, BS, 0, stream>>>(x, W1s, h1, N);
  k_zero<<<nb((long long)N * H1DIM), BS, 0, stream>>>(out1, N * H1DIM);
  k_attn1<<<nb((long long)N * NHEADS), BS, 0, stream>>>(h1, as1, ad1, s1, d1, amax1, den1, N);
  k_edge1_max<<<nb((long long)ET * NHEADS), BS, 0, stream>>>(ei, E, N, s1, d1, amax1);
  k_edge1_sum<<<nb((long long)ET * NHEADS), BS, 0, stream>>>(ei, E, N, s1, d1, amax1, den1);
  k_edge1_agg<<<nb((long long)ET * NHEADS), BS, 0, stream>>>(ei, E, N, s1, d1, amax1, den1, h1, out1);
  k_elu_bias1<<<nb((long long)N * H1DIM), BS, 0, stream>>>(out1, b1, h1p, N);

  // ----- layer 2 -----
  k_gemm2<<<gemmBlocks, BS, 0, stream>>>(h1p, W2s, h2, N);
  k_attn2<<<nb(N), BS, 0, stream>>>(h2, as2, ad2, s2, d2, amax2, den2, out, N);
  k_edge2_max<<<nb(ET), BS, 0, stream>>>(ei, E, N, s2, d2, amax2);
  k_edge2_sum<<<nb(ET), BS, 0, stream>>>(ei, E, N, s2, d2, amax2, den2);
  k_edge2_w<<<nb(ET), BS, 0, stream>>>(ei, E, N, s2, d2, amax2, den2, w2buf);
  k_edge2_agg<<<nb((long long)ET * OUTC), BS, 0, stream>>>(ei, E, N, w2buf, h2, out);
  k_elu_bias2<<<nb((long long)N * OUTC), BS, 0, stream>>>(out, b2, N);
}
